// Head_21311627723236
// MI455X (gfx1250) — matrix operand order repaired and checked by reading, not compiled, not run on hardware
//
#include <hip/hip_runtime.h>
#include <hip/hip_bf16.h>

#define B_ 4
#define T_ 4096
#define D_ 512
#define HS_ 64
#define NTILE 256            // T/16
#define NSEG 4               // j-segments for stats pass
#define SCALE 0.125f         // 1/sqrt(64)

typedef __attribute__((ext_vector_type(16))) __bf16          v16bf;
typedef __attribute__((ext_vector_type(8)))  unsigned int    v8ui;
typedef __attribute__((ext_vector_type(8)))  float           v8f;
typedef __attribute__((ext_vector_type(4)))  unsigned int    v4ui;
typedef __attribute__((ext_vector_type(8)))  int             v8i;
typedef __attribute__((ext_vector_type(4)))  int             v4i;

__device__ __forceinline__ unsigned short f32_to_bf16(float f) {
    unsigned int u = __float_as_uint(f);
    unsigned int r = (u + 0x7FFFu + ((u >> 16) & 1u)) >> 16;
    return (unsigned short)r;
}
__device__ __forceinline__ unsigned int pack_bf16(float lo, float hi) {
    return (unsigned int)f32_to_bf16(lo) | ((unsigned int)f32_to_bf16(hi) << 16);
}

__device__ __forceinline__ v8f wmma_bf16(v16bf a, v16bf b, v8f c) {
    return __builtin_amdgcn_wmma_f32_16x16x32_bf16(false, a, false, b, (short)0, c, false, false);
}

// A fragment (16 M x 32 K bf16): element (M,K) at base[M*ld + K]; dword loads.
__device__ __forceinline__ v16bf load_a_frag(const unsigned short* base, int ld, int lane) {
    const unsigned int* p = (const unsigned int*)base;
    int M = lane & 15, hi = lane >> 4;
    v8ui a;
#pragma unroll
    for (int g = 0; g < 8; ++g) {
        int K = ((g & 3) << 1) + ((g >> 2) << 4) + (hi << 3);   // even
        a[g] = p[(M * ld + K) >> 1];
    }
    return __builtin_bit_cast(v16bf, a);
}

// B fragment (32 K x 16 N bf16): element (K,N) at base[N*ld + K]; dword loads.
__device__ __forceinline__ v16bf load_b_frag(const unsigned short* base, int ld, int lane) {
    const unsigned int* p = (const unsigned int*)base;
    int N = lane & 15, hi = lane >> 4;
    v8ui b;
#pragma unroll
    for (int v = 0; v < 8; ++v) {
        int K = ((v & 3) << 1) + ((v >> 2) << 4) + (hi << 3);   // even
        b[v] = p[(N * ld + K) >> 1];
    }
    return __builtin_bit_cast(v16bf, b);
}

// ---------------------------------------------------------------------------
// TDM: async DMA of an int32 tile (tile_w=16 contiguous cols, tile_h rows,
// row stride T_) from global into LDS. D# per CDNA5 ISA.
// ---------------------------------------------------------------------------
__device__ __forceinline__ void tdm_load_i32_tile(const int* gptr, unsigned int lds_off,
                                                  int tile_h) {
    unsigned long long ga = (unsigned long long)(size_t)gptr;
    v4ui g0 = { 1u,                                   // count=1, user mode
                lds_off,                              // lds_addr (bytes)
                (unsigned int)ga,                     // global_addr[31:0]
                (unsigned int)((ga >> 32) & 0x01FFFFFFu) | (2u << 30) };  // addr[56:32]|type=2
    v8i g1 = { (int)0x00020000,                       // data_size=4B (code 2)
               (int)(((unsigned)T_ & 0xFFFFu) << 16), // tensor_dim0[15:0] @ [63:48]
               (int)(((unsigned)T_ & 0xFFFFu) << 16), // dim0[31:16]=0; tensor_dim1[15:0] @ [95:80]
               (int)(16u << 16),                      // dim1[31:16]=0; tile_dim0=16 @ [127:112]
               (int)(unsigned)tile_h,                 // tile_dim1 @ [143:128]; tile_dim2=0
               (int)T_,                               // tensor_dim0_stride[31:0]
               0, 0 };                                // stride hi / dim1_stride = 0
    v4i g2 = {0, 0, 0, 0};
    v4i g3 = {0, 0, 0, 0};
#if defined(__clang_major__) && __clang_major__ >= 23
    v8i gz = {0, 0, 0, 0, 0, 0, 0, 0};
    __builtin_amdgcn_tensor_load_to_lds(g0, g1, g2, g3, gz, 0);
#else
    __builtin_amdgcn_tensor_load_to_lds(g0, g1, g2, g3, 0);
#endif
}

// convert f32 [rows x cols] -> bf16 transposed [cols x rows]
__global__ void k_cvt_t(const float* __restrict__ src, unsigned short* __restrict__ dst,
                        int rows, int cols) {
    int i = blockIdx.x * blockDim.x + threadIdx.x;
    if (i < rows * cols) {
        int r = i / cols, c = i % cols;
        dst[c * rows + r] = f32_to_bf16(src[i]);
    }
}

// ---------------------------------------------------------------------------
// Projections: Q,K (bf16 [B,T,HS] row-major), V transposed (bf16 [B,HS,T]).
// B-fragment stream is software-pipelined one tile deep behind the WMMAs.
// ---------------------------------------------------------------------------
__global__ __launch_bounds__(128) void k_proj(
    const float* __restrict__ X,
    const unsigned short* __restrict__ Wqt,
    const unsigned short* __restrict__ Wkt,
    const unsigned short* __restrict__ Wvt,
    unsigned short* __restrict__ Qb,
    unsigned short* __restrict__ Kb,
    unsigned short* __restrict__ Vt)
{
    const int LDX = D_ + 8;
    __shared__ unsigned int XsmU[(16 * (D_ + 8)) / 2];
    unsigned short* Xsm = (unsigned short*)XsmU;

    int blk = blockIdx.x;
    int b   = blk >> 8;
    int t0  = (blk & 255) << 4;

    const float* Xrow = X + (size_t)(b * T_ + t0) * D_;
    for (int e = threadIdx.x; e < 16 * D_ / 2; e += 128) {
        int r = e >> 8, c = (e & 255) << 1;
        XsmU[(r * LDX + c) >> 1] = pack_bf16(Xrow[(size_t)r * D_ + c],
                                             Xrow[(size_t)r * D_ + c + 1]);
    }
    __syncthreads();

    int wave = threadIdx.x >> 5;
    int lane = threadIdx.x & 31;
    int nl   = lane & 15;
    int hi   = lane >> 4;

    for (int k = 0; k < 3; ++k) {
        int id = wave * 3 + k;                   // 0..11: (weight, h-tile)
        int w  = id >> 2;
        int h0 = (id & 3) << 4;
        const unsigned short* W = (w == 0) ? Wqt : (w == 1) ? Wkt : Wvt;

        v8f acc = {};
        v16bf bcur = load_b_frag(W, D_, lane);
#pragma unroll
        for (int d0 = 0; d0 < D_; d0 += 32) {
            v16bf bnxt = bcur;
            if (d0 + 32 < D_) bnxt = load_b_frag(W + d0 + 32, D_, lane);  // prefetch
            v16bf a = load_a_frag(Xsm + d0, LDX, lane);
            acc = wmma_bf16(a, bcur, acc);
            bcur = bnxt;
        }
        if (w < 2) {                             // Q,K row-major [t][h]
            unsigned short* O = (w == 0) ? Qb : Kb;
#pragma unroll
            for (int r = 0; r < 8; ++r) {
                int row = t0 + r + (hi << 3);
                O[(size_t)(b * T_ + row) * HS_ + h0 + nl] = f32_to_bf16(acc[r]);
            }
        } else {                                 // V transposed [h][t], packed stores
            unsigned int* O = (unsigned int*)Vt;
            size_t base = (size_t)(b * HS_ + h0 + nl) * T_ + t0 + (hi << 3);
#pragma unroll
            for (int r = 0; r < 8; r += 2)
                O[(base + r) >> 1] = pack_bf16(acc[r], acc[r + 1]);
        }
    }
}

// ---------------------------------------------------------------------------
// Pass 1a: partial softmax stats per (b, i-tile, j-segment).
// dag tiles arrive via double-buffered TDM (tensor_load_to_lds + tensorcnt).
// ---------------------------------------------------------------------------
__global__ __launch_bounds__(128) void k_stats(
    const unsigned short* __restrict__ Qb,
    const unsigned short* __restrict__ Kb,
    const int* __restrict__ dag,
    float* __restrict__ mP, float* __restrict__ zP)
{
    __shared__ int DagSm[4][2][16 * 16];         // per-wave double buffer

    int wave = threadIdx.x >> 5;
    int lane = threadIdx.x & 31;
    int unit = blockIdx.x * 4 + wave;            // 0..B*NTILE*NSEG-1
    int seg  = unit & (NSEG - 1);
    int tile = unit >> 2;                        // (b, itile)
    int b    = tile >> 8;
    int i0   = (tile & 255) << 4;
    int nl   = lane & 15;
    int hi   = lane >> 4;

    const float NEG = -__builtin_inff();

    const unsigned short* Qrow = Qb + (size_t)(b * T_ + i0) * HS_;
    v16bf aq0 = load_a_frag(Qrow,      HS_, lane);
    v16bf aq1 = load_a_frag(Qrow + 32, HS_, lane);

    float m[8], l[8];
#pragma unroll
    for (int r = 0; r < 8; ++r) { m[r] = NEG; l[r] = 0.f; }

    int jt0 = seg * (NTILE / NSEG), jt1 = jt0 + NTILE / NSEG;
    const int* dagBase = dag + (size_t)i0 * T_;
    unsigned int off0 = (unsigned int)(size_t)(void*)&DagSm[wave][0][0];
    unsigned int off1 = (unsigned int)(size_t)(void*)&DagSm[wave][1][0];

    tdm_load_i32_tile(dagBase + (jt0 << 4), off0, 16);
    int cur = 0;
    for (int jt = jt0; jt < jt1; ++jt) {
        int jn = (jt + 1 < jt1) ? jt + 1 : jt;
        tdm_load_i32_tile(dagBase + (jn << 4), cur ? off0 : off1, 16);
        int j0 = jt << 4;
        const unsigned short* Krow = Kb + (size_t)(b * T_ + j0) * HS_;
        __builtin_prefetch(Kb + (size_t)(b * T_ + (jn << 4) + (lane >> 1)) * HS_, 0, 0);
        v16bf bk0 = load_b_frag(Krow,      HS_, lane);
        v16bf bk1 = load_b_frag(Krow + 32, HS_, lane);
        v8f s = {};
        s = wmma_bf16(aq0, bk0, s);
        s = wmma_bf16(aq1, bk1, s);
        __builtin_amdgcn_s_wait_tensorcnt(1);    // current dag tile landed
        const int* dt = &DagSm[wave][cur][0];
#pragma unroll
        for (int r = 0; r < 8; ++r) {
            int dv   = dt[(r + (hi << 3)) * 16 + nl];
            float sv = dv ? (s[r] * SCALE) : NEG;
            float mo = m[r];
            float mn = fmaxf(mo, sv);
            if (mn > NEG) l[r] = l[r] * __expf(mo - mn) + __expf(sv - mn);
            m[r] = mn;
        }
        cur ^= 1;
    }
#pragma unroll
    for (int off = 8; off >= 1; off >>= 1) {
#pragma unroll
        for (int r = 0; r < 8; ++r) {
            float m2 = __shfl_xor(m[r], off, 32);
            float l2 = __shfl_xor(l[r], off, 32);
            float mn = fmaxf(m[r], m2);
            float ln = 0.f;
            if (mn > NEG) ln = l[r] * __expf(m[r] - mn) + l2 * __expf(m2 - mn);
            m[r] = mn; l[r] = ln;
        }
    }
    if (nl == 0) {
#pragma unroll
        for (int r = 0; r < 8; ++r) {
            int irow = i0 + r + (hi << 3);
            mP[seg * (B_ * T_) + b * T_ + irow] = m[r];
            zP[seg * (B_ * T_) + b * T_ + irow] = l[r];
        }
    }
}

// Pass 1b: merge the NSEG partial stats per row.
__global__ void k_merge(const float* __restrict__ mP, const float* __restrict__ zP,
                        float* __restrict__ mOut, float* __restrict__ zOut) {
    int i = blockIdx.x * blockDim.x + threadIdx.x;
    if (i >= B_ * T_) return;
    const float NEG = -__builtin_inff();
    float m = NEG;
#pragma unroll
    for (int s = 0; s < NSEG; ++s) m = fmaxf(m, mP[s * (B_ * T_) + i]);
    float z = 0.f;
    if (m > NEG) {
#pragma unroll
        for (int s = 0; s < NSEG; ++s) {
            float ms = mP[s * (B_ * T_) + i];
            if (ms > NEG) z += zP[s * (B_ * T_) + i] * __expf(ms - m);
        }
    }
    mOut[i] = m; zOut[i] = z;
}

// ---------------------------------------------------------------------------
// Pass 2: out[b,j,h] = swish( sum_i p[i,j] * V[i,h] ),  p = dag*exp(s-m)/Z.
// dag tiles via double-buffered TDM; P^T staged per-wave in LDS; partial
// 16x64 tiles merged across waves via ds_add_f32.
// ---------------------------------------------------------------------------
__global__ __launch_bounds__(128) void k_out(
    const unsigned short* __restrict__ Qb,
    const unsigned short* __restrict__ Kb,
    const unsigned short* __restrict__ Vt,
    const int* __restrict__ dag,
    const float* __restrict__ mArr,
    const float* __restrict__ zArr,
    float* __restrict__ out)
{
    __shared__ float        Osm[16 * 64];
    __shared__ unsigned int PsmU[4][16 * 20];    // per-wave P^T tile, stride 40 bf16
    __shared__ int          DagSm[4][2][32 * 16];

    int blk  = blockIdx.x;
    int b    = blk >> 8;
    int j0   = (blk & 255) << 4;
    int wave = threadIdx.x >> 5;
    int lane = threadIdx.x & 31;
    int nl   = lane & 15;
    int hi   = lane >> 4;

    for (int e = threadIdx.x; e < 1024; e += 128) Osm[e] = 0.f;
    __syncthreads();

    const unsigned short* Krow = Kb + (size_t)(b * T_ + j0) * HS_;
    v16bf bk0 = load_b_frag(Krow,      HS_, lane);
    v16bf bk1 = load_b_frag(Krow + 32, HS_, lane);

    v8f acc[4] = {v8f{}, v8f{}, v8f{}, v8f{}};
    unsigned int*   Pw = &PsmU[wave][0];
    unsigned short* Ps = (unsigned short*)Pw;
    unsigned int off0 = (unsigned int)(size_t)(void*)&DagSm[wave][0][0];
    unsigned int off1 = (unsigned int)(size_t)(void*)&DagSm[wave][1][0];

    tdm_load_i32_tile(dag + ((size_t)(wave << 5)) * T_ + j0, off0, 32);
    int cur = 0;
    for (int ch = wave; ch < T_ / 32; ch += 4) {
        int ibase = ch << 5;
        int chn = (ch + 4 < T_ / 32) ? ch + 4 : ch;
        tdm_load_i32_tile(dag + ((size_t)(chn << 5)) * T_ + j0, cur ? off0 : off1, 32);
        __builtin_prefetch(Qb + (size_t)(b * T_ + (chn << 5) + lane) * HS_, 0, 0);
        __builtin_amdgcn_s_wait_tensorcnt(1);    // current dag tile landed
        const int* dt = &DagSm[wave][cur][0];
#pragma unroll
        for (int it = 0; it < 2; ++it) {
            int irow0 = ibase + (it << 4);
            const unsigned short* Qrow = Qb + (size_t)(b * T_ + irow0) * HS_;
            v16bf aq0 = load_a_frag(Qrow,      HS_, lane);
            v16bf aq1 = load_a_frag(Qrow + 32, HS_, lane);
            v8f s = {};
            s = wmma_bf16(aq0, bk0, s);
            s = wmma_bf16(aq1, bk1, s);
#pragma unroll
            for (int r = 0; r < 8; r += 2) {
                float pv[2];
#pragma unroll
                for (int q = 0; q < 2; ++q) {
                    int il = (it << 4) + r + q + (hi << 3);      // 0..31 in chunk
                    int i  = ibase + il;
                    int dv = dt[il * 16 + nl];
                    float mi = mArr[b * T_ + i];
                    float zi = zArr[b * T_ + i];
                    pv[q] = 0.f;
                    if (dv && zi > 0.f) pv[q] = __expf(s[r + q] * SCALE - mi) / zi;
                }
                Pw[(nl * 40 + (it << 4) + (hi << 3) + r) >> 1] = pack_bf16(pv[0], pv[1]);
            }
        }
        v16bf ap = load_a_frag(Ps, 40, lane);    // P^T; same-wave DS is in-order
#pragma unroll
        for (int h = 0; h < 4; ++h) {
            const unsigned short* Vrow = Vt + (size_t)(b * HS_ + (h << 4)) * T_ + ibase;
            v16bf bv = load_b_frag(Vrow, T_, lane);
            acc[h] = wmma_bf16(ap, bv, acc[h]);
        }
        cur ^= 1;
    }

#pragma unroll
    for (int h = 0; h < 4; ++h)
#pragma unroll
        for (int r = 0; r < 8; ++r) {
            int jl = r + (hi << 3);
            atomicAdd(&Osm[jl * 64 + (h << 4) + nl], acc[h][r]);
        }
    __syncthreads();

    for (int e = threadIdx.x; e < 1024; e += 128) {
        int jl = e >> 6, hc = e & 63;
        float x = Osm[e];
        float y = x / (1.f + __expf(-x));        // swish
        out[(size_t)(b * T_ + j0 + jl) * HS_ + hc] = y;
    }
}

extern "C" void kernel_launch(void* const* d_in, const int* in_sizes, int n_in,
                              void* d_out, int out_size, void* d_ws, size_t ws_size,
                              hipStream_t stream) {
    const float* X   = (const float*)d_in[0];
    const int*   dag = (const int*)d_in[1];
    const float* Wk  = (const float*)d_in[2];
    const float* Wq  = (const float*)d_in[3];
    const float* Wv  = (const float*)d_in[4];
    float* out = (float*)d_out;

    unsigned short* Wqt = (unsigned short*)d_ws;            // [HS][D]
    unsigned short* Wkt = Wqt + D_ * HS_;
    unsigned short* Wvt = Wkt + D_ * HS_;
    unsigned short* Qb  = Wvt + D_ * HS_;                   // [B][T][HS]
    unsigned short* Kb  = Qb + (size_t)B_ * T_ * HS_;       // [B][T][HS]
    unsigned short* Vt  = Kb + (size_t)B_ * T_ * HS_;       // [B][HS][T]
    float* mP   = (float*)(Vt + (size_t)B_ * T_ * HS_);     // [NSEG][B*T]
    float* zP   = mP + NSEG * B_ * T_;
    float* mArr = zP + NSEG * B_ * T_;
    float* zArr = mArr + B_ * T_;

    const int nW = D_ * HS_;
    k_cvt_t<<<(nW + 255) / 256, 256, 0, stream>>>(Wq, Wqt, D_, HS_);
    k_cvt_t<<<(nW + 255) / 256, 256, 0, stream>>>(Wk, Wkt, D_, HS_);
    k_cvt_t<<<(nW + 255) / 256, 256, 0, stream>>>(Wv, Wvt, D_, HS_);

    k_proj <<<B_ * NTILE, 128, 0, stream>>>(X, Wqt, Wkt, Wvt, Qb, Kb, Vt);
    k_stats<<<B_ * NTILE * NSEG / 4, 128, 0, stream>>>(Qb, Kb, dag, mP, zP);
    k_merge<<<(B_ * T_ + 255) / 256, 256, 0, stream>>>(mP, zP, mArr, zArr);
    k_out  <<<B_ * NTILE, 128, 0, stream>>>(Qb, Kb, Vt, dag, mArr, zArr, out);
}
